// RecurrentResourceActionHead_74998718922890
// MI455X (gfx1250) — matrix operand order repaired and checked by reading, not compiled, not run on hardware
//
#include <hip/hip_runtime.h>

typedef __attribute__((ext_vector_type(16))) _Float16 v16h;
typedef __attribute__((ext_vector_type(8)))  _Float16 v8h;
typedef __attribute__((ext_vector_type(4)))  _Float16 v4h;
typedef __attribute__((ext_vector_type(8)))  float    v8f;
typedef __attribute__((ext_vector_type(4)))  unsigned int v4u;
typedef __attribute__((ext_vector_type(8)))  int      v8i;
typedef __attribute__((ext_vector_type(4)))  int      v4i;

#define BSZ   131072
#define DM    256
#define RR    6
#define HH    128
#define CNT   4
#define NEGV  (-1000000000.0f)

#define ROWS_PER_BLOCK 64
#define NBLOCKS (BSZ / ROWS_PER_BLOCK)   // 2048

// LDS layout (bytes)
#define XSTR  264        // X tile: f16 elems per row (256 + 8 pad)
#define W2STR 132        // W2 LDS tile: f16 elems per row (128 + 4, via TDM pad)
#define HSTR  136        // H tile: f16 elems per row (128 + 8 pad)
#define XW2_OFF  0       // Xh f16 [64][264] (33792 B) -> then Wt2 f16 [128][132] (33792 B)
#define HH_OFF   33792   // Hh f16 [64][136]  (17408 B)
#define WD_OFF   51200   // Wd f32 [128][6]   (3072 B)
#define BD_OFF   54272   // bd f32 [6] (pad 32)
#define OUTS_OFF 54304   // output state f32 [64][8] (2048 B)
#define ENT_OFF  56352   // ent partials f32 [64] (256 B)
#define SMEM_BYTES 56608

// -------- prep: transpose + f32->f16 convert W1[:256] and W2 into d_ws --------
__global__ void prep_weights(const float* __restrict__ W1, const float* __restrict__ W2,
                             _Float16* __restrict__ wt1, _Float16* __restrict__ wt2) {
  int idx = blockIdx.x * blockDim.x + threadIdx.x;
  if (idx < HH * DM) {                 // wt1[n][k] = W1[k][n]
    int n = idx / DM, k = idx % DM;
    wt1[idx] = (_Float16)W1[k * HH + n];
  }
  int i2 = idx - HH * DM;
  if (i2 >= 0 && i2 < HH * HH) {       // wt2[n][k] = W2[k][n]
    int n = i2 / HH, k = i2 % HH;
    wt2[i2] = (_Float16)W2[k * HH + n];
  }
}

// -------- deterministic final reduction of entropy partials --------
__global__ void ent_reduce(const float* __restrict__ part, float* __restrict__ dst) {
  if (threadIdx.x == 0 && blockIdx.x == 0) {
    float s = 0.f;
    for (int k = 0; k < NBLOCKS; ++k) s += part[k];
    dst[0] = s * (1.0f / (float)BSZ);
  }
}

// -------- main fused kernel: 64 rows per block, 4 waves --------
__global__ void __launch_bounds__(128)
rrah_main(const float* __restrict__ mainin, const float* __restrict__ cres,
          const int*   __restrict__ acts,   const float* __restrict__ htm,
          const int*   __restrict__ pha,
          const float* __restrict__ W1,     const float* __restrict__ b1,
          const float* __restrict__ gammav, const float* __restrict__ betav,
          const float* __restrict__ b2,     const float* __restrict__ Wd,
          const float* __restrict__ bd,
          const _Float16* __restrict__ wt1, const _Float16* __restrict__ wt2,
          float* __restrict__ out, float* __restrict__ entpart) {
  __shared__ __align__(16) unsigned char smem[SMEM_BYTES];
  _Float16* Xh    = (_Float16*)(smem + XW2_OFF);
  _Float16* Wt2L  = (_Float16*)(smem + XW2_OFF);
  _Float16* Hh    = (_Float16*)(smem + HH_OFF);
  float*    wds   = (float*)(smem + WD_OFF);
  float*    bds   = (float*)(smem + BD_OFF);
  float*    outS  = (float*)(smem + OUTS_OFF);
  float*    entb  = (float*)(smem + ENT_OFF);

  const int tid  = threadIdx.x;
  const int wave = tid >> 5;
  const int lane = tid & 31;
  const int halfw= lane >> 4;
  const int mcol = lane & 15;
  const int rw0  = wave * 16;
  const int row0 = blockIdx.x * ROWS_PER_BLOCK;

  // ---- cooperative loads: small weights + state init ----
  for (int q = tid; q < HH * RR; q += 128) wds[q] = Wd[q];
  if (tid < RR) bds[tid] = bd[tid];
  for (int q = tid; q < ROWS_PER_BLOCK * 8; q += 128) outS[q] = 0.f;

  // ---- per-lane column-constant registers ----
  float gv[8], btv[8], b2v[8], b1v[8], w1c[RR][8];
  #pragma unroll
  for (int nt = 0; nt < 8; ++nt) {
    int col = nt * 16 + mcol;
    gv[nt] = gammav[col]; btv[nt] = betav[col];
    b2v[nt] = b2[col];    b1v[nt] = b1[col];
    #pragma unroll
    for (int j = 0; j < RR; ++j) w1c[j][nt] = W1[DM * HH + j * HH + col];
  }

  // ---- load main_inputs tile [64][256] f32 -> f16 into Xh ----
  {
    const float4* src = (const float4*)(mainin + (size_t)row0 * DM);
    for (int v = tid; v < ROWS_PER_BLOCK * DM / 4; v += 128) {
      float4 f = src[v];
      int flat = v * 4, r = flat / DM, c = flat % DM;
      _Float16* d = Xh + r * XSTR + c;
      d[0] = (_Float16)f.x; d[1] = (_Float16)f.y;
      d[2] = (_Float16)f.z; d[3] = (_Float16)f.w;
    }
  }
  __syncthreads();

  // ---- GEMM1: base = main @ W1[:256] + b1  (kept in registers per wave) ----
  v8f acc[8];
  #pragma unroll
  for (int nt = 0; nt < 8; ++nt)
    #pragma unroll
    for (int e = 0; e < 8; ++e) acc[nt][e] = 0.0f;
  {
    const _Float16* ap = Xh + (rw0 + mcol) * XSTR;
    for (int k0 = 0; k0 < DM; k0 += 32) {
      v8h lo = *(const v8h*)(ap + k0 + halfw * 16);
      v8h hi = *(const v8h*)(ap + k0 + halfw * 16 + 8);
      v16h a;
      #pragma unroll
      for (int e = 0; e < 8; ++e) { a[e] = lo[e]; a[e + 8] = hi[e]; }
      #pragma unroll
      for (int nt = 0; nt < 8; ++nt) {
        v16h bfr = *(const v16h*)(wt1 + (size_t)(nt * 16 + mcol) * DM + k0 + halfw * 16);
        acc[nt] = __builtin_amdgcn_wmma_f32_16x16x32_f16(
            false, a, false, bfr, (short)0, acc[nt], false, false);
      }
    }
  }
  #pragma unroll
  for (int nt = 0; nt < 8; ++nt)
    #pragma unroll
    for (int e = 0; e < 8; ++e) acc[nt][e] += b1v[nt];
  __syncthreads();   // all waves done reading Xh; region becomes Wt2L

  // ---- TDM: pull W2 (f16, 32 KB) global -> LDS with pad to 132-elem rows ----
  if (__builtin_amdgcn_readfirstlane(tid) == 0) {    // wave 0 only (uniform branch)
    unsigned long long ga = (unsigned long long)(uintptr_t)wt2;
    unsigned int ldsa = (unsigned int)(uintptr_t)(smem + XW2_OFF);
    v4u g0;
    g0[0] = 1u;                                     // count=1 valid descriptor
    g0[1] = ldsa;                                   // lds_addr
    g0[2] = (unsigned int)(ga & 0xffffffffu);       // global_addr lo
    g0[3] = (unsigned int)((ga >> 32) & 0x1ffffffu) | 0x80000000u; // hi | type=2
    v8i g1;
    g1[0] = 0x03510000;  // data_size=2B | pad_en | pad_interval=64dw | pad_amount=2dw
    g1[1] = 0x40000000;  // tensor_dim0 = 16384 (bits 79:48 low half)
    g1[2] = 0x00010000;  // tensor_dim1 = 1
    g1[3] = 0x40000000;  // tile_dim0 = 16384
    g1[4] = 0x00000001;  // tile_dim1 = 1
    g1[5] = 16384;       // tensor_dim0_stride
    g1[6] = 0; g1[7] = 0;
    v4i g2; g2[0] = 0; g2[1] = 0; g2[2] = 0; g2[3] = 0;
    v4i g3; g3[0] = 0; g3[1] = 0; g3[2] = 0; g3[3] = 0;
    v8i g4; 
    #pragma unroll
    for (int e = 0; e < 8; ++e) g4[e] = 0;
    __builtin_amdgcn_tensor_load_to_lds(g0, g1, g2, g3, g4, 0);
    __builtin_amdgcn_s_wait_tensorcnt((short)0);
  }
  __syncthreads();

  // ---- per-row recurrent state (thread t owns row t, t < 64) ----
  float cr[RR], maskv[RR];
  float lp_acc = 0.f, ent_acc = 0.f;
  if (tid < ROWS_PER_BLOCK) {
    float s = 0.f;
    #pragma unroll
    for (int j = 0; j < RR; ++j) {
      float c = cres[(size_t)(row0 + tid) * RR + j];
      cr[j] = c; s += c; maskv[j] = (c > 0.f) ? 1.f : 0.f;
    }
    maskv[0] = (s == 0.f) ? 1.f : 0.f;
  }

  for (int step = 0; step < CNT; ++step) {
    // ---- phase T (wave layout): pre = base + output@W1tail ; LN ; ReLU -> Hh ----
    #pragma unroll
    for (int r = 0; r < 8; ++r) {
      int row = rw0 + halfw * 8 + r;
      float o[RR];
      #pragma unroll
      for (int j = 0; j < RR; ++j) o[j] = outS[row * 8 + j];
      float pr[8]; float s = 0.f, q = 0.f;
      #pragma unroll
      for (int nt = 0; nt < 8; ++nt) {
        float v = acc[nt][r];
        #pragma unroll
        for (int j = 0; j < RR; ++j) v += o[j] * w1c[j][nt];
        pr[nt] = v; s += v; q += v * v;
      }
      #pragma unroll
      for (int m = 1; m < 16; m <<= 1) { s += __shfl_xor(s, m, 32); q += __shfl_xor(q, m, 32); }
      float mu = s * (1.f / HH);
      float rs = rsqrtf(q * (1.f / HH) - mu * mu + 1e-5f);
      _Float16* hrow = Hh + row * HSTR;
      #pragma unroll
      for (int nt = 0; nt < 8; ++nt) {
        float ln = (pr[nt] - mu) * rs * gv[nt] + btv[nt];
        hrow[nt * 16 + mcol] = (_Float16)fmaxf(ln, 0.f);
      }
    }
    // no barrier: each wave reads back only its own 16 rows (DS in-order per wave)

    // ---- GEMM2: h = relu_ln @ W2 (WMMA, B from LDS), +b2 -> Hh (f16) ----
    v8f c2[8];
    #pragma unroll
    for (int nt = 0; nt < 8; ++nt)
      #pragma unroll
      for (int e = 0; e < 8; ++e) c2[nt][e] = 0.0f;
    {
      const _Float16* ap = Hh + (rw0 + mcol) * HSTR;
      for (int k0 = 0; k0 < HH; k0 += 32) {
        v8h lo = *(const v8h*)(ap + k0 + halfw * 16);
        v8h hi = *(const v8h*)(ap + k0 + halfw * 16 + 8);
        v16h a;
        #pragma unroll
        for (int e = 0; e < 8; ++e) { a[e] = lo[e]; a[e + 8] = hi[e]; }
        #pragma unroll
        for (int nt = 0; nt < 8; ++nt) {
          const _Float16* bp = Wt2L + (nt * 16 + mcol) * W2STR + k0 + halfw * 16;
          v4h q0 = *(const v4h*)(bp);     v4h q1 = *(const v4h*)(bp + 4);
          v4h q2 = *(const v4h*)(bp + 8); v4h q3 = *(const v4h*)(bp + 12);
          v16h bfr;
          #pragma unroll
          for (int e = 0; e < 4; ++e) {
            bfr[e] = q0[e]; bfr[4 + e] = q1[e]; bfr[8 + e] = q2[e]; bfr[12 + e] = q3[e];
          }
          c2[nt] = __builtin_amdgcn_wmma_f32_16x16x32_f16(
              false, a, false, bfr, (short)0, c2[nt], false, false);
        }
      }
    }
    #pragma unroll
    for (int nt = 0; nt < 8; ++nt)
      #pragma unroll
      for (int r = 0; r < 8; ++r)
        Hh[(rw0 + halfw * 8 + r) * HSTR + nt * 16 + mcol] = (_Float16)(c2[nt][r] + b2v[nt]);
    __syncthreads();

    // ---- phase T2 (thread per row): logits, masked log-softmax, bookkeeping ----
    if (tid < ROWS_PER_BLOCK) {
      const int row = row0 + tid;
      float lg[RR];
      #pragma unroll
      for (int j = 0; j < RR; ++j) lg[j] = bds[j];
      const _Float16* hrow = Hh + tid * HSTR;
      for (int n = 0; n < HH; ++n) {
        float h = (float)hrow[n];
        #pragma unroll
        for (int j = 0; j < RR; ++j) lg[j] += h * wds[n * RR + j];
      }
      float ml[RR];
      #pragma unroll
      for (int j = 0; j < RR; ++j) ml[j] = (maskv[j] > 0.f) ? lg[j] : NEGV;
      int mode = 0; float mx = ml[0];
      #pragma unroll
      for (int j = 1; j < RR; ++j) if (ml[j] > mx) { mx = ml[j]; mode = j; }
      float se = 0.f;
      #pragma unroll
      for (int j = 0; j < RR; ++j) se += expf(ml[j] - mx);
      float lse = mx + logf(se);
      float ent = 0.f;
      #pragma unroll
      for (int j = 0; j < RR; ++j) {
        float lp = ml[j] - lse;
        float p  = expf(lp);
        ent -= p * lp;
      }
      int a = acts[(size_t)row * CNT + step];
      float lp_a = ml[a] - lse;
      if (step > 0) {
        float g = (acts[(size_t)row * CNT + step - 1] > 0) ? 1.f : 0.f;
        lp_a *= g; ent *= g;
      }
      lp_acc += lp_a; ent_acc += ent;
      out[(size_t)BSZ * RR + (size_t)row * CNT + step] = (float)mode;
      outS[tid * 8 + a] += 1.f; outS[tid * 8 + 0] = 0.f;
      cr[a] = fmaxf(cr[a] - 1.f, 0.f);
      #pragma unroll
      for (int j = 0; j < RR; ++j) maskv[j] = (cr[j] > 0.f) ? 1.f : 0.f;
      maskv[0] = 1.f;
    }
    __syncthreads();
  }

  // ---- outputs ----
  if (tid < ROWS_PER_BLOCK) {
    const int row = row0 + tid;
    #pragma unroll
    for (int j = 0; j < RR; ++j) out[(size_t)row * RR + j] = outS[tid * 8 + j];
    float hmv = htm[pha[row]];
    out[(size_t)BSZ * 10 + row] = lp_acc * hmv;
    entb[tid] = ent_acc * hmv;
  }
  __syncthreads();
  if (tid == 0) {
    float s = 0.f;
    for (int k = 0; k < ROWS_PER_BLOCK; ++k) s += entb[k];   // fixed order: deterministic
    entpart[blockIdx.x] = s;
  }
}

extern "C" void kernel_launch(void* const* d_in, const int* in_sizes, int n_in,
                              void* d_out, int out_size, void* d_ws, size_t ws_size,
                              hipStream_t stream) {
  const float* mainin = (const float*)d_in[0];
  const float* cresp  = (const float*)d_in[1];
  const int*   actsp  = (const int*)  d_in[2];
  const float* htmp   = (const float*)d_in[3];
  const int*   phap   = (const int*)  d_in[4];
  const float* W1p    = (const float*)d_in[5];
  const float* b1p    = (const float*)d_in[6];
  const float* gp     = (const float*)d_in[7];
  const float* btp    = (const float*)d_in[8];
  const float* W2p    = (const float*)d_in[9];
  const float* b2p    = (const float*)d_in[10];
  const float* Wdp    = (const float*)d_in[11];
  const float* bdp    = (const float*)d_in[12];
  float* outp = (float*)d_out;

  _Float16* wt1 = (_Float16*)d_ws;                       // 128*256 f16 = 64 KiB
  _Float16* wt2 = wt1 + HH * DM;                         // 128*128 f16 = 32 KiB
  float* entpart = (float*)((char*)d_ws + 98304);        // 2048 f32

  prep_weights<<<(HH * DM + HH * HH + 255) / 256, 256, 0, stream>>>(W1p, W2p, wt1, wt2);
  rrah_main<<<NBLOCKS, 128, 0, stream>>>(mainin, cresp, actsp, htmp, phap,
                                         W1p, b1p, gp, btp, b2p, Wdp, bdp,
                                         wt1, wt2, outp, entpart);
  ent_reduce<<<1, 32, 0, stream>>>(entpart, outp + (size_t)BSZ * 11);
}
